// CombinedModel_86887188398804
// MI455X (gfx1250) — matrix operand order repaired and checked by reading, not compiled, not run on hardware
//
#include <hip/hip_runtime.h>

typedef __attribute__((ext_vector_type(16))) _Float16 v16h;
typedef __attribute__((ext_vector_type(8)))  _Float16 v8h;
typedef __attribute__((ext_vector_type(4)))  _Float16 v4h;
typedef __attribute__((ext_vector_type(8)))  float    v8f;

#define Bdim 32
#define Sdim 2048
#define Hdim 768
#define FMAXdim 32
#define MS 64
#define NH 12
#define DH 64
#define Ldim 96
#define FFNdim 3072

#define WAVES 8
#define NT 6                 // 16-wide n-tiles per wave
#define MT 2                 // 16-row m-tiles per block (B-fragment reuse x2)
#define BM (MT*16)           // 32 rows per block
#define BN (WAVES*NT*16)     // 768 columns per block
#define KSTEP 64
#define LDA_SH 72            // padded halfs per LDS A row (144B, 16B aligned, conflict-free)

enum { EPI_BIAS=0, EPI_SCORE=1, EPI_LN=2, EPI_GELU=3 };

// ---------------------------------------------------------------------------
// Fused WMMA GEMM: C[M,N] = epi(A[M,K] (f16, row stride lda) @ Bt^T)
// Bt pre-transposed f16: Bt[n*K+k] = W[k,n] -> lane B fragment = 16 contiguous
// halfs (contiguous k = 16h + s order, matching A). Block = 32 rows x 768 cols; every B fragment
// feeds MT=2 WMMAs (halves L2 weight traffic). A double-buffered in LDS, one
// barrier per KSTEP=64; stage-store placed after compute so its load-wait
// cannot stall the WMMA stream.
// ---------------------------------------------------------------------------
template<int EPI>
__global__ __launch_bounds__(256)
void gemm_wmma(const _Float16* __restrict__ A, long lda,
               const _Float16* __restrict__ Bt, int K,
               const float* __restrict__ bias,
               float* __restrict__ C, _Float16* __restrict__ Ch, long ldc,
               const float* __restrict__ res, long ldres,
               const float* __restrict__ ln_g, const float* __restrict__ ln_b,
               const float* __restrict__ vp, float* __restrict__ scoreOut,
               int Ncols)
{
  __shared__ _Float16 As[2][BM * LDA_SH];
  __shared__ float red[2 * BM];

  const int tid  = threadIdx.x;
  const int wave = tid >> 5;
  const int lane = tid & 31;
  const long mbase = (long)blockIdx.x * BM;
  const long nbase = (long)blockIdx.y * BN + (long)wave * (NT * 16);
  const int  lo16  = (lane < 16);

  if (EPI == EPI_SCORE || EPI == EPI_LN) {
    if (tid < 2 * BM) red[tid] = 0.0f;
  }

  // staging: 32 rows x 64 halfs = 2048 halfs; 256 threads x 8 halfs (16B)
  const int se = tid * 8;
  const int sr = se >> 6;          // row 0..31
  const int sk = se & 63;          // k offset 0..56

  // prologue: stage k0 = 0 into buffer 0
  *(v8h*)&As[0][sr * LDA_SH + sk] = *(const v8h*)&A[(mbase + sr) * lda + sk];
  __syncthreads();

  v8f acc[MT][NT] = {};
  const int arow = lane & 15;
  const int akb  = lo16 ? 0 : 16;
  const int bkb  = lo16 ? 0 : 16;
  int buf = 0;

  for (int k0 = 0; k0 < K; k0 += KSTEP) {
    const bool has_next = (k0 + KSTEP) < K;
    v8h stagev = {};
    if (has_next)
      stagev = *(const v8h*)&A[(mbase + sr) * lda + (k0 + KSTEP + sk)];

#pragma unroll
    for (int sub = 0; sub < 2; ++sub) {
      // A fragments for both m-tiles (contiguous k = 16h + s lane layout)
      v16h afrag[MT];
#pragma unroll
      for (int mt = 0; mt < MT; ++mt) {
        const _Float16* ab = &As[buf][(mt * 16 + arow) * LDA_SH + sub * 32 + akb];
        v8h alo = *(const v8h*)ab;
        v8h ahi = *(const v8h*)(ab + 8);
#pragma unroll
        for (int i = 0; i < 8; ++i) { afrag[mt][i] = alo[i]; afrag[mt][8 + i] = ahi[i]; }
      }

      const int kbB = k0 + sub * 32 + bkb;
#pragma unroll
      for (int t = 0; t < NT; ++t) {
        const long n = nbase + t * 16 + (lane & 15);
        const v8h* bp = (const v8h*)(Bt + n * (long)K + kbB);
        v8h blo = bp[0], bhi = bp[1];
        v16h bfr;
#pragma unroll
        for (int i = 0; i < 8; ++i) { bfr[i] = blo[i]; bfr[8 + i] = bhi[i]; }
#pragma unroll
        for (int mt = 0; mt < MT; ++mt)   // reuse each B fragment for 2 WMMAs
          acc[mt][t] = __builtin_amdgcn_wmma_f32_16x16x32_f16(
              false, afrag[mt], false, bfr, (short)0, acc[mt][t], false, false);
      }
    }

    if (has_next)
      *(v8h*)&As[buf ^ 1][sr * LDA_SH + sk] = stagev;
    __syncthreads();
    buf ^= 1;
  }

  const int rlo = lo16 ? 0 : 8;  // acc[mt][t][j] = C[mt*16 + rlo + j][nbase + t*16 + (lane&15)]

  if (EPI == EPI_BIAS) {
#pragma unroll
    for (int t = 0; t < NT; ++t) {
      const long n = nbase + t * 16 + (lane & 15);
      const float bn = bias[n];
#pragma unroll
      for (int mt = 0; mt < MT; ++mt)
#pragma unroll
        for (int j = 0; j < 8; ++j)
          C[(mbase + mt * 16 + rlo + j) * ldc + n] = acc[mt][t][j] + bn;
    }
  } else if (EPI == EPI_GELU) {   // f16 out (feeds next GEMM's A)
#pragma unroll
    for (int t = 0; t < NT; ++t) {
      const long n = nbase + t * 16 + (lane & 15);
      const float bn = bias[n];
#pragma unroll
      for (int mt = 0; mt < MT; ++mt)
#pragma unroll
        for (int j = 0; j < 8; ++j) {
          float val = acc[mt][t][j] + bn;
          Ch[(mbase + mt * 16 + rlo + j) * ldc + n] =
              (_Float16)(0.5f * val * (1.0f + erff(val * 0.70710678118654752f)));
        }
    }
  } else if (EPI == EPI_SCORE) {
#pragma unroll
    for (int mt = 0; mt < MT; ++mt)
#pragma unroll
      for (int j = 0; j < 8; ++j) {
        float p = 0.0f;
#pragma unroll
        for (int t = 0; t < NT; ++t) {
          const long n = nbase + t * 16 + (lane & 15);
          p += tanhf(acc[mt][t][j] + bias[n]) * vp[n];
        }
        atomicAdd(&red[mt * 16 + rlo + j], p);
      }
    __syncthreads();
    if (tid < BM) scoreOut[mbase + tid] = red[tid];
  } else { // EPI_LN : bias + residual + LayerNorm over full row (grid.y == 1)
#pragma unroll
    for (int mt = 0; mt < MT; ++mt)
#pragma unroll
      for (int j = 0; j < 8; ++j) {
        float s = 0.0f, s2 = 0.0f;
#pragma unroll
        for (int t = 0; t < NT; ++t) {
          const long n = nbase + t * 16 + (lane & 15);
          float val = acc[mt][t][j] + bias[n] +
                      res[(mbase + mt * 16 + rlo + j) * ldres + n];
          acc[mt][t][j] = val;
          s += val; s2 += val * val;
        }
        atomicAdd(&red[mt * 16 + rlo + j], s);
        atomicAdd(&red[BM + mt * 16 + rlo + j], s2);
      }
    __syncthreads();
#pragma unroll
    for (int mt = 0; mt < MT; ++mt)
#pragma unroll
      for (int j = 0; j < 8; ++j) {
        const int r = mt * 16 + rlo + j;
        const float mu  = red[r] / (float)Ncols;
        const float var = red[BM + r] / (float)Ncols - mu * mu;
        const float rs  = rsqrtf(var + 1e-12f);
#pragma unroll
        for (int t = 0; t < NT; ++t) {
          const long n = nbase + t * 16 + (lane & 15);
          const float y = (acc[mt][t][j] - mu) * rs * ln_g[n] + ln_b[n];
          C[(mbase + r) * ldc + n] = y;
          if (Ch) Ch[(mbase + r) * ldc + n] = (_Float16)y;
        }
      }
  }
}

// --------------------------------------------------------------------------
// dst[n*Kd + k] = (f16) src[k*Nd + n]   (weight transpose + fp32->f16)
// --------------------------------------------------------------------------
__global__ void transpose_convert(const float* __restrict__ src,
                                  _Float16* __restrict__ dst, int Kd, int Nd) {
  long i = (long)blockIdx.x * 256 + threadIdx.x;
  if (i < (long)Kd * Nd) {
    long n = i / Kd, k = i % Kd;
    dst[i] = (_Float16)src[k * (long)Nd + n];
  }
}

// --------------------------------------------------------------------------
// flat fp32 -> f16 (vectorized x4)
// --------------------------------------------------------------------------
__global__ void convert_f16(const float* __restrict__ src,
                            _Float16* __restrict__ dst, long n4) {
  long i = (long)blockIdx.x * 256 + threadIdx.x;
  if (i < n4) {
    float4 v = ((const float4*)src)[i];
    v4h o;
    o[0] = (_Float16)v.x; o[1] = (_Float16)v.y;
    o[2] = (_Float16)v.z; o[3] = (_Float16)v.w;
    ((v4h*)dst)[i] = o;
  }
}

// --------------------------------------------------------------------------
// Per-batch SEP scan: seg_id (-1 invalid), n_sent, n_eff, tot = n_eff + fc
// --------------------------------------------------------------------------
__global__ void segment_kernel(const int* __restrict__ ids,
                               const int* __restrict__ facts_count,
                               const int* __restrict__ sep_id_p,
                               int* __restrict__ seg_id,
                               int* __restrict__ n_sent_o,
                               int* __restrict__ n_eff_o,
                               int* __restrict__ tot_o) {
  int b = threadIdx.x;
  if (b >= Bdim) return;
  const int sep = sep_id_p[0];
  const int* row = ids + (long)b * Sdim;
  int cnt = 0;
  for (int s = 0; s < Sdim; ++s) cnt += (row[s] == sep) ? 1 : 0;
  const int n_sent = cnt;
  int seg = 0;
  for (int s = 0; s < Sdim; ++s) {
    const bool is_sep = (row[s] == sep);
    const bool valid = (s >= 1) && !is_sep && (seg < n_sent);
    seg_id[(long)b * Sdim + s] = valid ? seg : -1;
    if (is_sep) seg++;
  }
  const int ne = n_sent > 1 ? n_sent : 1;
  n_sent_o[b] = n_sent;
  n_eff_o[b] = ne;
  tot_o[b] = ne + facts_count[b];
}

// --------------------------------------------------------------------------
// Segment softmax (per batch): scores -> e = exp(score - segmax); zinv = 1/z
// --------------------------------------------------------------------------
__global__ __launch_bounds__(256)
void seg_softmax_kernel(float* __restrict__ scores, const int* __restrict__ seg_id,
                        float* __restrict__ zinv) {
  __shared__ unsigned mbits[MS];
  __shared__ float ssum[MS];
  const int b = blockIdx.x, tid = threadIdx.x;
  if (tid < MS) { mbits[tid] = 0u; ssum[tid] = 0.0f; }
  __syncthreads();
  for (int s = tid; s < Sdim; s += 256) {
    const int sg = seg_id[(long)b * Sdim + s];
    if (sg >= 0) {
      unsigned u = __float_as_uint(scores[(long)b * Sdim + s]);
      u = (u & 0x80000000u) ? ~u : (u | 0x80000000u);
      atomicMax(&mbits[sg], u);
    }
  }
  __syncthreads();
  for (int s = tid; s < Sdim; s += 256) {
    const long idx = (long)b * Sdim + s;
    const int sg = seg_id[idx];
    float e = 0.0f;
    if (sg >= 0) {
      const unsigned u = mbits[sg];
      const unsigned bits = (u & 0x80000000u) ? (u ^ 0x80000000u) : ~u;
      e = expf(scores[idx] - __uint_as_float(bits));
      atomicAdd(&ssum[sg], e);
    }
    scores[idx] = e;
  }
  __syncthreads();
  if (tid < MS) {
    const float z = ssum[tid];
    zinv[b * MS + tid] = 1.0f / (z > 0.0f ? z : 1.0f);
  }
}

// --------------------------------------------------------------------------
// sent[b,s,:] = sum_tok e * hidden1 * zinv ; override sent[b,0] if n_sent==0
// --------------------------------------------------------------------------
__global__ __launch_bounds__(128)
void sent_agg_kernel(const float* __restrict__ hidden1, const float* __restrict__ e,
                     const int* __restrict__ seg_id, const float* __restrict__ zinv,
                     const int* __restrict__ n_sent, float* __restrict__ sent) {
  __shared__ float accs[MS * 128];
  const int b = blockIdx.x, hc = blockIdx.y, tid = threadIdx.x;
  const int h = hc * 128 + tid;
  for (int sgv = 0; sgv < MS; ++sgv) accs[sgv * 128 + tid] = 0.0f;
  __syncthreads();
  const float* hb = hidden1 + (long)b * Sdim * Hdim;
  for (int s = 0; s < Sdim; ++s) {
    const int sg = seg_id[(long)b * Sdim + s];
    if (sg >= 0) {
      const float w = e[(long)b * Sdim + s];
      accs[sg * 128 + tid] += w * hb[(long)s * Hdim + h];
    }
  }
  __syncthreads();
  for (int sgv = 0; sgv < MS; ++sgv)
    sent[((long)b * MS + sgv) * Hdim + h] = accs[sgv * 128 + tid] * zinv[b * MS + sgv];
  if (n_sent[b] == 0)
    sent[((long)b * MS) * Hdim + h] = hb[h];
}

// --------------------------------------------------------------------------
// x[b,l,:] = sent | fact_cls | 0 ; writes fp32 (residual) + f16 (GEMM A)
// --------------------------------------------------------------------------
__global__ void build_x_kernel(const float* __restrict__ sent,
                               const float* __restrict__ fact_cls,
                               const int* __restrict__ n_eff,
                               const int* __restrict__ facts_count,
                               float* __restrict__ x, _Float16* __restrict__ xh) {
  long i = (long)blockIdx.x * 256 + threadIdx.x;
  if (i >= (long)Bdim * Ldim * Hdim) return;
  const int h = (int)(i % Hdim);
  const long t = i / Hdim;
  const int l = (int)(t % Ldim);
  const int b = (int)(t / Ldim);
  const int ne = n_eff[b], fc = facts_count[b];
  float v = 0.0f;
  if (l < ne)           v = sent[((long)b * MS + l) * Hdim + h];
  else if (l < ne + fc) v = fact_cls[((long)b * FMAXdim + (l - ne)) * Hdim + h];
  x[i] = v;
  xh[i] = (_Float16)v;
}

// --------------------------------------------------------------------------
// Attention, query l=0 only. block = (b, head), 128 threads. ctx written f16.
// --------------------------------------------------------------------------
__global__ __launch_bounds__(128)
void attn_kernel(const float* __restrict__ q, const float* __restrict__ k,
                 const float* __restrict__ v, const int* __restrict__ tot,
                 _Float16* __restrict__ ctx) {
  __shared__ float qv[DH];
  __shared__ float sc[Ldim];
  __shared__ float mz[2];
  const int bh = blockIdx.x, b = bh / NH, h = bh % NH, tid = threadIdx.x;
  const float* kb = k + (long)b * Ldim * Hdim + h * DH;
  const float* vb = v + (long)b * Ldim * Hdim + h * DH;
  if (tid < DH) qv[tid] = q[(long)b * Hdim + h * DH + tid];
  __syncthreads();
  if (tid < Ldim) {
    float s = 0.0f;
    for (int d = 0; d < DH; ++d) s += qv[d] * kb[(long)tid * Hdim + d];
    s *= 0.125f;                         // 1/sqrt(64)
    if (tid >= tot[b]) s -= 1e9f;
    sc[tid] = s;
  }
  __syncthreads();
  if (tid == 0) {
    float m = -3.4e38f;
    for (int i = 0; i < Ldim; ++i) m = fmaxf(m, sc[i]);
    mz[0] = m;
  }
  __syncthreads();
  if (tid < Ldim) sc[tid] = expf(sc[tid] - mz[0]);
  __syncthreads();
  if (tid == 0) {
    float z = 0.0f;
    for (int i = 0; i < Ldim; ++i) z += sc[i];
    mz[1] = z;
  }
  __syncthreads();
  if (tid < DH) {
    float c = 0.0f;
    for (int m = 0; m < Ldim; ++m) c += sc[m] * vb[(long)m * Hdim + tid];
    ctx[(long)b * Hdim + h * DH + tid] = (_Float16)(c / mz[1]);
  }
}

// ===========================================================================
extern "C" void kernel_launch(void* const* d_in, const int* in_sizes, int n_in,
                              void* d_out, int out_size, void* d_ws, size_t ws_size,
                              hipStream_t stream) {
  const float* hidden1  = (const float*)d_in[0];
  const float* fact_cls = (const float*)d_in[1];
  const float* Wp  = (const float*)d_in[2];
  const float* bp  = (const float*)d_in[3];
  const float* vp  = (const float*)d_in[4];
  const float* Wq  = (const float*)d_in[5];
  const float* bq  = (const float*)d_in[6];
  const float* Wk  = (const float*)d_in[7];
  const float* bk  = (const float*)d_in[8];
  const float* Wv  = (const float*)d_in[9];
  const float* bv  = (const float*)d_in[10];
  const float* Wo  = (const float*)d_in[11];
  const float* bo  = (const float*)d_in[12];
  const float* ln1g = (const float*)d_in[13];
  const float* ln1b = (const float*)d_in[14];
  const float* Wf1 = (const float*)d_in[15];
  const float* bf1 = (const float*)d_in[16];
  const float* Wf2 = (const float*)d_in[17];
  const float* bf2 = (const float*)d_in[18];
  const float* ln2g = (const float*)d_in[19];
  const float* ln2b = (const float*)d_in[20];
  const int* input_ids   = (const int*)d_in[21];
  const int* facts_count = (const int*)d_in[22];
  const int* sep_id_p    = (const int*)d_in[24];

  char* wsp = (char*)d_ws;
  auto alloc = [&](size_t bytes) -> void* {
    void* p = (void*)wsp;
    wsp += (bytes + 255) & ~(size_t)255;
    return p;
  };
  _Float16* WpT  = (_Float16*)alloc((size_t)Hdim * Hdim * 2);
  _Float16* WqT  = (_Float16*)alloc((size_t)Hdim * Hdim * 2);
  _Float16* WkT  = (_Float16*)alloc((size_t)Hdim * Hdim * 2);
  _Float16* WvT  = (_Float16*)alloc((size_t)Hdim * Hdim * 2);
  _Float16* WoT  = (_Float16*)alloc((size_t)Hdim * Hdim * 2);
  _Float16* Wf1T = (_Float16*)alloc((size_t)Hdim * FFNdim * 2);
  _Float16* Wf2T = (_Float16*)alloc((size_t)FFNdim * Hdim * 2);
  _Float16* h1h  = (_Float16*)alloc((size_t)Bdim * Sdim * Hdim * 2);   // hidden1 f16
  float* scores = (float*)alloc((size_t)Bdim * Sdim * 4);
  int*   seg_id = (int*)alloc((size_t)Bdim * Sdim * 4);
  float* zinv   = (float*)alloc((size_t)Bdim * MS * 4);
  int* n_sent_a = (int*)alloc(Bdim * 4);
  int* n_eff_a  = (int*)alloc(Bdim * 4);
  int* tot_a    = (int*)alloc(Bdim * 4);
  float* sent = (float*)alloc((size_t)Bdim * MS * Hdim * 4);
  float* xbuf = (float*)alloc((size_t)Bdim * Ldim * Hdim * 4);
  _Float16* xh = (_Float16*)alloc((size_t)Bdim * Ldim * Hdim * 2);
  float* qbuf = (float*)alloc((size_t)Bdim * Hdim * 4);
  float* kbuf = (float*)alloc((size_t)Bdim * Ldim * Hdim * 4);
  float* vbuf = (float*)alloc((size_t)Bdim * Ldim * Hdim * 4);
  _Float16* ch  = (_Float16*)alloc((size_t)Bdim * Hdim * 2);
  float* x1     = (float*)alloc((size_t)Bdim * Hdim * 4);
  _Float16* x1h = (_Float16*)alloc((size_t)Bdim * Hdim * 2);
  _Float16* hh  = (_Float16*)alloc((size_t)Bdim * FFNdim * 2);

  // 1) weight transpose+convert, hidden1 fp32 -> f16
  {
    int gHH = (Hdim * Hdim + 255) / 256;
    int gHF = (Hdim * FFNdim + 255) / 256;
    transpose_convert<<<gHH, 256, 0, stream>>>(Wp, WpT, Hdim, Hdim);
    transpose_convert<<<gHH, 256, 0, stream>>>(Wq, WqT, Hdim, Hdim);
    transpose_convert<<<gHH, 256, 0, stream>>>(Wk, WkT, Hdim, Hdim);
    transpose_convert<<<gHH, 256, 0, stream>>>(Wv, WvT, Hdim, Hdim);
    transpose_convert<<<gHH, 256, 0, stream>>>(Wo, WoT, Hdim, Hdim);
    transpose_convert<<<gHF, 256, 0, stream>>>(Wf1, Wf1T, Hdim, FFNdim);
    transpose_convert<<<gHF, 256, 0, stream>>>(Wf2, Wf2T, FFNdim, Hdim);
    long n4 = (long)Bdim * Sdim * Hdim / 4;
    convert_f16<<<(int)((n4 + 255) / 256), 256, 0, stream>>>(hidden1, h1h, n4);
  }

  // 2) SEP segmentation
  segment_kernel<<<1, 32, 0, stream>>>(input_ids, facts_count, sep_id_p,
                                       seg_id, n_sent_a, n_eff_a, tot_a);

  // 3) scores = tanh(hidden1 @ Wp + bp) . vp  (fused WMMA GEMM + row reduction)
  gemm_wmma<EPI_SCORE><<<dim3((Bdim * Sdim) / BM, 1), 256, 0, stream>>>(
      h1h, Hdim, WpT, Hdim, bp, nullptr, nullptr, 0, nullptr, 0,
      nullptr, nullptr, vp, scores, Hdim);

  // 4) segment softmax, 5) weighted segment-sum -> sent (fp32 precision)
  seg_softmax_kernel<<<Bdim, 256, 0, stream>>>(scores, seg_id, zinv);
  sent_agg_kernel<<<dim3(Bdim, Hdim / 128), 128, 0, stream>>>(
      hidden1, scores, seg_id, zinv, n_sent_a, sent);

  // 6) assemble x (fp32 residual + f16 GEMM operand)
  build_x_kernel<<<(int)(((long)Bdim * Ldim * Hdim + 255) / 256), 256, 0, stream>>>(
      sent, fact_cls, n_eff_a, facts_count, xbuf, xh);

  // 7) projections: q only for l=0 (row stride L*H picks x[b,0,:]); k,v all rows
  gemm_wmma<EPI_BIAS><<<dim3(Bdim / BM, 1), 256, 0, stream>>>(
      xh, (long)Ldim * Hdim, WqT, Hdim, bq, qbuf, nullptr, Hdim,
      nullptr, 0, nullptr, nullptr, nullptr, nullptr, Hdim);
  gemm_wmma<EPI_BIAS><<<dim3((Bdim * Ldim) / BM, 1), 256, 0, stream>>>(
      xh, Hdim, WkT, Hdim, bk, kbuf, nullptr, Hdim,
      nullptr, 0, nullptr, nullptr, nullptr, nullptr, Hdim);
  gemm_wmma<EPI_BIAS><<<dim3((Bdim * Ldim) / BM, 1), 256, 0, stream>>>(
      xh, Hdim, WvT, Hdim, bv, vbuf, nullptr, Hdim,
      nullptr, 0, nullptr, nullptr, nullptr, nullptr, Hdim);

  // 8) attention (query row 0 only), ctx emitted f16
  attn_kernel<<<Bdim * NH, 128, 0, stream>>>(qbuf, kbuf, vbuf, tot_a, ch);

  // 9) x1 = LN1(x[:,0] + ctx @ Wo + bo) ; also emit f16 copy for FFN1
  gemm_wmma<EPI_LN><<<dim3(Bdim / BM, 1), 256, 0, stream>>>(
      ch, Hdim, WoT, Hdim, bo, x1, x1h, Hdim,
      xbuf, (long)Ldim * Hdim, ln1g, ln1b, nullptr, nullptr, Hdim);

  // 10) h = gelu(x1 @ Wf1 + bf1)  (f16 out; N=3072 -> 4 column chunks)
  gemm_wmma<EPI_GELU><<<dim3(Bdim / BM, FFNdim / BN), 256, 0, stream>>>(
      x1h, Hdim, Wf1T, Hdim, bf1, nullptr, hh, FFNdim,
      nullptr, 0, nullptr, nullptr, nullptr, nullptr, FFNdim);

  // 11) out = LN2(x1 + h @ Wf2 + bf2)
  gemm_wmma<EPI_LN><<<dim3(Bdim / BM, 1), 256, 0, stream>>>(
      hh, FFNdim, Wf2T, FFNdim, bf2, (float*)d_out, nullptr, Hdim,
      x1, Hdim, ln2g, ln2b, nullptr, nullptr, Hdim);

  (void)in_sizes; (void)n_in; (void)out_size; (void)ws_size;
}
